// Attention_org_27307402068381
// MI455X (gfx1250) — matrix operand order repaired and checked by reading, not compiled, not run on hardware
//
#include <hip/hip_runtime.h>
#include <hip/hip_bf16.h>
#include <math.h>

typedef __attribute__((ext_vector_type(16))) _Float16 v16h;
typedef __attribute__((ext_vector_type(8)))  _Float16 v8h;
typedef __attribute__((ext_vector_type(8)))  float    v8f;
typedef __attribute__((ext_vector_type(4)))  int      v4i;

union U16 { v16h v; v8h h[2]; };

#define INORM_EPS 1e-5f
#define AS1 __attribute__((address_space(1)))
#define AS3 __attribute__((address_space(3)))

#if defined(__HIP_DEVICE_COMPILE__) && __has_builtin(__builtin_amdgcn_global_load_async_to_lds_b128)
#define USE_ASYNC_LDS 1
#else
#define USE_ASYNC_LDS 0
#endif

static constexpr int Bdim = 8;
static constexpr int Ndim = 1024;
static constexpr int Hdim = 4;
static constexpr int KVd  = 960;

__device__ __forceinline__ void wait_async_zero() {
#if USE_ASYNC_LDS
#if __has_builtin(__builtin_amdgcn_s_wait_asynccnt)
    __builtin_amdgcn_s_wait_asynccnt(0);
#else
    asm volatile("s_wait_asynccnt 0" ::: "memory");
#endif
#endif
}

// ---------------------------------------------------------------------------
// f32 -> f16 conversion
// ---------------------------------------------------------------------------
__global__ void cvt_f32_f16(const float* __restrict__ in, _Float16* __restrict__ out, long n) {
    long i = (long)blockIdx.x * blockDim.x + threadIdx.x;
    if (i < n) out[i] = (_Float16)in[i];
}

// ---------------------------------------------------------------------------
// Generic batched, segmented-K, LDS-tiled WMMA GEMM.
// D[m,n] = alpha * sum_s sum_k A[s][m,k] * Bt[s][n,k]
// A tile rows = M dim, B tile rows = N dim, contraction along last index.
// Workgroup tile 64x64, 256 threads (8 wave32), Kc = 32 per step.
// Double-buffered LDS staged via GLOBAL_LOAD_ASYNC_TO_LDS_B128 (ASYNCcnt),
// one workgroup barrier per K-step, incrementally advanced source pointers.
// ---------------------------------------------------------------------------
template<bool TRANSD, bool HALFOUT>
__global__ __launch_bounds__(256) void gemm16(
    const _Float16* __restrict__ A, const _Float16* __restrict__ Bt, void* __restrict__ D,
    int lda, int ldb, int ldd,
    int K, int kseg, long segA, long segB,
    int aDiv, long aBatch, int bMod, long bBatch, long dBatch,
    float alpha)
{
    __shared__ _Float16 As[2][64 * 40];
    __shared__ _Float16 Bs[2][64 * 40];

    const int z = blockIdx.z;
    const _Float16* Ab = A  + (long)(z / aDiv) * aBatch;
    const _Float16* Bb = Bt + (long)(z % bMod) * bBatch;
    const long dOff = (long)z * dBatch;

    const int m0 = blockIdx.y * 64;
    const int n0 = blockIdx.x * 64;

    const int t    = threadIdx.x;
    const int lane = t & 31;
    const int wave = t >> 5;
    const int tm   = wave & 3;          // which 16-row tile of the 64-row block
    const int tn   = (wave >> 2) * 2;   // two adjacent 16-col tiles per wave

    v8f acc0 = {};
    v8f acc1 = {};

    const int srow = t >> 2;            // staging row 0..63
    const int skc  = (t & 3) * 8;       // staging k-chunk 0,8,16,24 (halves)

    const int kchunks = K >> 5;         // K is always a multiple of 32 here
    const int total   = kseg * kchunks;

    // Incrementally advanced per-thread source pointers (next chunk to issue).
    const _Float16* pA = Ab + (long)(m0 + srow) * lda + skc;
    const _Float16* pB = Bb + (long)(n0 + srow) * ldb + skc;
    const long segStepA = segA - (long)(kchunks - 1) * 32;
    const long segStepB = segB - (long)(kchunks - 1) * 32;
    int kcnt = 0;                       // chunk index within current segment

    auto advance = [&]() {
        const bool wrap = (kcnt == kchunks - 1);
        pA += wrap ? segStepA : 32;
        pB += wrap ? segStepB : 32;
        kcnt = wrap ? 0 : kcnt + 1;
    };

#if USE_ASYNC_LDS
    AS3 v4i* const ldsA0 = (AS3 v4i*)&As[0][srow * 40 + skc];
    AS3 v4i* const ldsA1 = (AS3 v4i*)&As[1][srow * 40 + skc];
    AS3 v4i* const ldsB0 = (AS3 v4i*)&Bs[0][srow * 40 + skc];
    AS3 v4i* const ldsB1 = (AS3 v4i*)&Bs[1][srow * 40 + skc];
    auto issue = [&](int buf) {
        __builtin_amdgcn_global_load_async_to_lds_b128(
            (AS1 v4i*)pA, buf ? ldsA1 : ldsA0, 0, 0);
        __builtin_amdgcn_global_load_async_to_lds_b128(
            (AS1 v4i*)pB, buf ? ldsB1 : ldsB0, 0, 0);
    };
#else
    auto issue = [&](int buf) {
        *reinterpret_cast<float4*>(&As[buf][srow * 40 + skc]) =
            *reinterpret_cast<const float4*>(pA);
        *reinterpret_cast<float4*>(&Bs[buf][srow * 40 + skc]) =
            *reinterpret_cast<const float4*>(pB);
    };
#endif

    // prologue: fill buffer 0 with chunk 0
    issue(0);
    advance();

    for (int c = 0; c < total; ++c) {
        const int buf = c & 1;
        wait_async_zero();          // own async loads for `buf` complete
        __syncthreads();            // tile visible to every wave
        if (c + 1 < total) {
            issue(buf ^ 1);
            advance();
        }

        // ---- A fragment (16x32 f16): lanes 0-15 -> K 0..15, lanes 16-31 -> K 16..31
        const int lm  = tm * 16 + (lane & 15);
        const int kbA = (lane >> 4) * 16;
        U16 af;
        af.h[0] = *reinterpret_cast<const v8h*>(&As[buf][lm * 40 + kbA]);
        af.h[1] = *reinterpret_cast<const v8h*>(&As[buf][lm * 40 + kbA + 8]);

        // ---- B fragments (32x16 f16): lanes 0-15 -> K 0..15, lanes 16-31 -> K 16..31
        const int kbB = (lane >> 4) * 16;
        const int ln0 = tn * 16 + (lane & 15);
        U16 bf0, bf1;
        bf0.h[0] = *reinterpret_cast<const v8h*>(&Bs[buf][ln0 * 40 + kbB]);
        bf0.h[1] = *reinterpret_cast<const v8h*>(&Bs[buf][ln0 * 40 + kbB + 8]);
        const int ln1 = ln0 + 16;
        bf1.h[0] = *reinterpret_cast<const v8h*>(&Bs[buf][ln1 * 40 + kbB]);
        bf1.h[1] = *reinterpret_cast<const v8h*>(&Bs[buf][ln1 * 40 + kbB + 8]);

        acc0 = __builtin_amdgcn_wmma_f32_16x16x32_f16(false, af.v, false, bf0.v,
                                                      (short)0, acc0, false, false);
        acc1 = __builtin_amdgcn_wmma_f32_16x16x32_f16(false, af.v, false, bf1.v,
                                                      (short)0, acc1, false, false);
#if !USE_ASYNC_LDS
        __syncthreads();            // sync path needs the trailing barrier
#endif
    }

    // ---- epilogue: D layout — VGPR r: M = r + 8*(lane>=16); N = lane&15
    const int mmBase = m0 + tm * 16 + ((lane >> 4) << 3);
    const int nnA    = n0 + tn * 16 + (lane & 15);
    const int nnB    = nnA + 16;
    for (int r = 0; r < 8; ++r) {
        const int mm = mmBase + r;
        const float v0 = acc0[r] * alpha;
        const float v1 = acc1[r] * alpha;
        const long i0 = TRANSD ? ((long)nnA * ldd + mm) : ((long)mm * ldd + nnA);
        const long i1 = TRANSD ? ((long)nnB * ldd + mm) : ((long)mm * ldd + nnB);
        if (HALFOUT) {
            reinterpret_cast<_Float16*>(D)[dOff + i0] = (_Float16)v0;
            reinterpret_cast<_Float16*>(D)[dOff + i1] = (_Float16)v1;
        } else {
            reinterpret_cast<float*>(D)[dOff + i0] = v0;
            reinterpret_cast<float*>(D)[dOff + i1] = v1;
        }
    }
}

// ---------------------------------------------------------------------------
// Instance-norm stats: one block per (b,h) plane of C*KV f32 scores.
// ---------------------------------------------------------------------------
__global__ __launch_bounds__(256) void inorm_stats(const float* __restrict__ scores,
                                                   float* __restrict__ stats, int plane) {
    __shared__ float red[256];
    const long base = (long)blockIdx.x * plane;
    float s = 0.f, s2 = 0.f;
    for (int i = threadIdx.x; i < plane; i += 256) {
        const float x = scores[base + i];
        s += x; s2 += x * x;
    }
    red[threadIdx.x] = s; __syncthreads();
    for (int st = 128; st > 0; st >>= 1) {
        if (threadIdx.x < st) red[threadIdx.x] += red[threadIdx.x + st];
        __syncthreads();
    }
    const float sum = red[0]; __syncthreads();
    red[threadIdx.x] = s2; __syncthreads();
    for (int st = 128; st > 0; st >>= 1) {
        if (threadIdx.x < st) red[threadIdx.x] += red[threadIdx.x + st];
        __syncthreads();
    }
    if (threadIdx.x == 0) {
        const float sumsq = red[0];
        const float mu  = sum / (float)plane;
        const float var = sumsq / (float)plane - mu * mu;
        stats[blockIdx.x * 2 + 0] = mu;
        stats[blockIdx.x * 2 + 1] = rsqrtf(var + INORM_EPS);
    }
}

// ---------------------------------------------------------------------------
// Normalized softmax over last axis (KV=960): one block per (b,h,c) row.
// ---------------------------------------------------------------------------
__global__ __launch_bounds__(256) void inorm_softmax(const float* __restrict__ scores,
                                                     const float* __restrict__ stats,
                                                     _Float16* __restrict__ probs, int C) {
    __shared__ float red[256];
    const int row = blockIdx.x;
    const int z   = row / C;                   // (b,h)
    const long base = (long)row * KVd;
    const float mu = stats[z * 2 + 0];
    const float rs = stats[z * 2 + 1];

    float y[4];
    float m = -3.4e38f;
    for (int j = 0; j < 4; ++j) {
        const int idx = threadIdx.x + j * 256;
        if (idx < KVd) {
            const float x = (scores[base + idx] - mu) * rs;
            y[j] = x;
            m = fmaxf(m, x);
        } else {
            y[j] = -3.4e38f;
        }
    }
    red[threadIdx.x] = m; __syncthreads();
    for (int st = 128; st > 0; st >>= 1) {
        if (threadIdx.x < st) red[threadIdx.x] = fmaxf(red[threadIdx.x], red[threadIdx.x + st]);
        __syncthreads();
    }
    m = red[0]; __syncthreads();

    float s = 0.f;
    for (int j = 0; j < 4; ++j) {
        const int idx = threadIdx.x + j * 256;
        if (idx < KVd) {
            y[j] = __expf(y[j] - m);
            s += y[j];
        }
    }
    red[threadIdx.x] = s; __syncthreads();
    for (int st = 128; st > 0; st >>= 1) {
        if (threadIdx.x < st) red[threadIdx.x] += red[threadIdx.x + st];
        __syncthreads();
    }
    const float inv = 1.f / red[0];
    for (int j = 0; j < 4; ++j) {
        const int idx = threadIdx.x + j * 256;
        if (idx < KVd) probs[base + idx] = (_Float16)(y[j] * inv);
    }
}

// ---------------------------------------------------------------------------
// Host-side orchestration
// ---------------------------------------------------------------------------
extern "C" void kernel_launch(void* const* d_in, const int* in_sizes, int n_in,
                              void* d_out, int out_size, void* d_ws, size_t ws_size,
                              hipStream_t stream) {
    (void)in_sizes; (void)n_in; (void)out_size; (void)ws_size;
    const int Cs[4] = {64, 128, 256, 512};
    const int Cmax  = 512;

    // ---- workspace layout (256B aligned)
    char* w = (char*)d_ws;
    auto alloc = [&](size_t bytes) -> void* {
        void* p = w;
        w += (bytes + 255) & ~(size_t)255;
        return p;
    };
    _Float16* embAllH = (_Float16*)alloc((size_t)Bdim * Ndim * KVd * 2);
    _Float16* WkH     = (_Float16*)alloc((size_t)Hdim * KVd * KVd * 2);
    _Float16* WvH     = (_Float16*)alloc((size_t)Hdim * KVd * KVd * 2);
    _Float16* KtH     = (_Float16*)alloc((size_t)Bdim * Hdim * KVd * Ndim * 2); // [b,h,j,n]
    _Float16* VH      = (_Float16*)alloc((size_t)Bdim * Hdim * Ndim * KVd * 2); // [b,h,n,j]
    _Float16* QtH     = (_Float16*)alloc((size_t)Bdim * Hdim * Cmax * Ndim * 2); // [b,h,d,n]
    float*    scoresF = (float*)   alloc((size_t)Bdim * Hdim * Cmax * KVd * 4);  // [b,h,c,j]
    float*    statsF  = (float*)   alloc((size_t)Bdim * Hdim * 2 * 4);
    _Float16* probsH  = (_Float16*)alloc((size_t)Bdim * Hdim * Cmax * KVd * 2);  // [b,h,c,j]
    _Float16* ctxH    = (_Float16*)alloc((size_t)Bdim * Ndim * Cmax * 2);        // [b,n,c]
    _Float16* embH    = (_Float16*)alloc((size_t)Bdim * Ndim * Cmax * 2);
    _Float16* WqH     = (_Float16*)alloc((size_t)Hdim * Cmax * Cmax * 2);
    _Float16* WoH     = (_Float16*)alloc((size_t)Cmax * Cmax * 2);

    auto cvt = [&](const void* in, _Float16* out, long n) {
        const int blocks = (int)((n + 255) / 256);
        cvt_f32_f16<<<blocks, 256, 0, stream>>>((const float*)in, out, n);
    };

    // ---- shared K/V path
    cvt(d_in[4],  embAllH, (long)Bdim * Ndim * KVd);
    cvt(d_in[9],  WkH,     (long)Hdim * KVd * KVd);
    cvt(d_in[10], WvH,     (long)Hdim * KVd * KVd);

    // K[b,h,n,j] = sum_k embAll[b,n,k] * Wk[h,j,k]; stored transposed -> Kt[b,h,j,n]
    {
        dim3 g(KVd / 64, Ndim / 64, Bdim * Hdim);
        gemm16<true, true><<<g, 256, 0, stream>>>(
            embAllH, WkH, KtH,
            /*lda*/KVd, /*ldb*/KVd, /*ldd*/Ndim, /*K*/KVd, 1, 0, 0,
            /*aDiv*/Hdim, (long)Ndim * KVd, /*bMod*/Hdim, (long)KVd * KVd,
            (long)KVd * Ndim, 1.0f);
        // V stored natural -> V[b,h,n,j]
        gemm16<false, true><<<g, 256, 0, stream>>>(
            embAllH, WvH, VH,
            KVd, KVd, /*ldd*/KVd, KVd, 1, 0, 0,
            Hdim, (long)Ndim * KVd, Hdim, (long)KVd * KVd,
            (long)Ndim * KVd, 1.0f);
    }

    long outOff = 0;
    for (int br = 0; br < 4; ++br) {
        const int c = Cs[br];

        cvt(d_in[br],      embH, (long)Bdim * Ndim * c);
        cvt(d_in[5 + br],  WqH,  (long)Hdim * c * c);
        cvt(d_in[11 + br], WoH,  (long)c * c);

        // Q[b,h,n,d] = sum_k emb[b,n,k] * Wq[h,d,k]; stored transposed -> Qt[b,h,d,n]
        {
            dim3 g(c / 64, Ndim / 64, Bdim * Hdim);
            gemm16<true, true><<<g, 256, 0, stream>>>(
                embH, WqH, QtH,
                /*lda*/c, /*ldb*/c, /*ldd*/Ndim, /*K*/c, 1, 0, 0,
                /*aDiv*/Hdim, (long)Ndim * c, /*bMod*/Hdim, (long)c * c,
                (long)c * Ndim, 1.0f);
        }

        // scores[b,h,c,j] = (1/sqrt(960)) * sum_n Qt[b,h,c,n] * Kt[b,h,j,n]
        {
            dim3 g(KVd / 64, c / 64, Bdim * Hdim);
            gemm16<false, false><<<g, 256, 0, stream>>>(
                QtH, KtH, scoresF,
                /*lda*/Ndim, /*ldb*/Ndim, /*ldd*/KVd, /*K*/Ndim, 1, 0, 0,
                /*aDiv*/1, (long)c * Ndim, /*bMod*/Bdim * Hdim, (long)KVd * Ndim,
                (long)c * KVd, 0.032274861218395141f);
        }

        // instance-norm stats + softmax -> probs f16
        inorm_stats<<<Bdim * Hdim, 256, 0, stream>>>(scoresF, statsF, c * KVd);
        inorm_softmax<<<Bdim * Hdim * c, 256, 0, stream>>>(scoresF, statsF, probsH, c);

        // ctxMean[b,n,c] = (1/H) * sum_h sum_j probs[b,h,c,j] * V[b,h,n,j]
        // (segmented K over h, transposed f16 store)
        {
            dim3 g(Ndim / 64, c / 64, Bdim);
            gemm16<true, true><<<g, 256, 0, stream>>>(
                probsH, VH, ctxH,
                /*lda*/KVd, /*ldb*/KVd, /*ldd*/c, /*K*/KVd, /*kseg*/Hdim,
                /*segA*/(long)c * KVd, /*segB*/(long)Ndim * KVd,
                /*aDiv*/1, (long)Hdim * c * KVd, /*bMod*/Bdim, (long)Hdim * Ndim * KVd,
                (long)Ndim * c, 0.25f);
        }

        // out[b,n,e] = sum_c ctxMean[b,n,c] * Wo[e,c]  (f32 straight into d_out)
        {
            dim3 g(c / 64, Ndim / 64, Bdim);
            gemm16<false, false><<<g, 256, 0, stream>>>(
                ctxH, WoH, (float*)d_out + outOff,
                /*lda*/c, /*ldb*/c, /*ldd*/c, /*K*/c, 1, 0, 0,
                /*aDiv*/1, (long)Ndim * c, /*bMod*/1, 0,
                (long)Ndim * c, 1.0f);
        }
        outOff += (long)Bdim * Ndim * c;
    }
}
